// SelfAttention3D_31061203484909
// MI455X (gfx1250) — matrix operand order repaired and checked by reading, not compiled, not run on hardware
//
#include <hip/hip_runtime.h>

typedef unsigned short u16;
typedef __attribute__((ext_vector_type(8)))  __bf16 v8bf;
typedef __attribute__((ext_vector_type(16))) __bf16 v16bf;
typedef __attribute__((ext_vector_type(8)))  float  v8f;

#define S_TOT 4128           // 4 mem tokens + 28 masked pad + 4096 real tokens
#define SCALE 0.17677669529663687f   // 32^-0.5

// ---------------- helpers ----------------

__device__ __forceinline__ u16 f2bf(float f) {           // f32 -> bf16 RNE
    unsigned u = __float_as_uint(f);
    u += 0x7FFFu + ((u >> 16) & 1u);
    return (u16)(u >> 16);
}

__device__ __forceinline__ v8f vzero8() {
    v8f z = {0.f, 0.f, 0.f, 0.f, 0.f, 0.f, 0.f, 0.f};
    return z;
}

__device__ __forceinline__ v16bf cat16(v8bf lo, v8bf hi) {
    return __builtin_shufflevector(lo, hi, 0,1,2,3,4,5,6,7,8,9,10,11,12,13,14,15);
}

// A fragment (16x32 bf16, M x K) from a per-lane pointer.
// Lane pointer = Arow + (row0 + (lane&15))*pitch + k0 + (lane>>4)*8.
// ISA layout: elem e -> K = (e<8?0:16) + kh*8 + (e&7)  => two 16B loads at +0 and +16 elems.
__device__ __forceinline__ v16bf load_a_ptr(const u16* p) {
    v8bf lo = *(const v8bf*)p;
    v8bf hi = *(const v8bf*)(p + 16);
    return cat16(lo, hi);
}

// B fragment (32x16 bf16, K x N) from Bt storage (Bt[n][k] = B[k][n]) per-lane pointer.
// Lane pointer = Bt + (n0 + (lane&15))*pitch + k0 + (lane>>4)*8.
// ISA layout: elem e -> K = (e<8?0:16) + kh*8 + (e&7)  => two 16B loads at +0 and +16 elems.
__device__ __forceinline__ v16bf load_b_ptr(const u16* p) {
    v8bf lo = *(const v8bf*)p;
    v8bf hi = *(const v8bf*)(p + 16);
    return cat16(lo, hi);
}

__device__ __forceinline__ v8f wmma_bf16(v16bf a, v16bf b, v8f c) {
    return __builtin_amdgcn_wmma_f32_16x16x32_bf16(false, a, false, b, (short)0, c, false, false);
}

// ---------------- prep kernels ----------------

// x (b,c,h,w) f32 -> Xt[b][p][c] bf16
__global__ void prep_x_kernel(const float* __restrict__ x, u16* __restrict__ Xt) {
    int i = blockIdx.x * 256 + threadIdx.x;       // 4*256*4096 threads
    int p = i & 4095;
    int c = (i >> 12) & 255;
    int b = i >> 20;
    float v = x[((size_t)(b * 256 + c)) * 4096 + p];
    Xt[((size_t)(b * 4096 + p)) * 256 + c] = f2bf(v);
}

__global__ void cvt_bf16_kernel(const float* __restrict__ in, u16* __restrict__ out, int n) {
    int i = blockIdx.x * 256 + threadIdx.x;
    if (i < n) out[i] = f2bf(in[i]);
}

// mem_kv (2,8,4,32) -> Kb rows 0..31 (4..31 zero), Vt cols 0..31 (4..31 zero)
__global__ void prep_mem_kernel(const float* __restrict__ mem_kv,
                                u16* __restrict__ Kb, u16* __restrict__ Vt) {
    int i = blockIdx.x * 256 + threadIdx.x;       // 32*32*32 threads
    int d = i & 31;
    int s = (i >> 5) & 31;
    int bh = i >> 10;                              // b*8 + head
    int head = bh & 7;
    u16 kv = 0, vv = 0;
    if (s < 4) {
        kv = f2bf(mem_kv[((size_t)((0 * 8 + head) * 4 + s)) * 32 + d]);
        vv = f2bf(mem_kv[((size_t)((1 * 8 + head) * 4 + s)) * 32 + d]);
    }
    Kb[((size_t)bh * S_TOT + s) * 32 + d] = kv;
    Vt[((size_t)bh * 32 + d) * S_TOT + s] = vv;
}

// ---------------- QKV projection GEMM (2x2 register-blocked bf16 WMMA) ----------------

__device__ __forceinline__ void qkv_store_tile(v8f acc, const float* __restrict__ bqkv,
                                               int b, int mbase, int obase, int m, int kh,
                                               u16* __restrict__ Qb, u16* __restrict__ Kb,
                                               u16* __restrict__ Vt) {
    int o = obase + m;                             // wave-uniform sel/head per sub-tile
    float bias = bqkv[o];
    int sel  = o >> 8;                             // 0=q 1=k 2=v
    int head = (o >> 5) & 7;
    int d    = o & 31;
    int bh   = b * 8 + head;
#pragma unroll
    for (int r = 0; r < 8; r++) {
        int p = mbase + kh * 8 + r;                // C-frag row
        float v = acc[r] + bias;
        if (sel == 0)      Qb[((size_t)bh * 4096 + p) * 32 + d] = f2bf(v * SCALE);
        else if (sel == 1) Kb[((size_t)bh * S_TOT + 32 + p) * 32 + d] = f2bf(v);
        else               Vt[((size_t)bh * 32 + d) * S_TOT + 32 + p] = f2bf(v);
    }
}

// C[p,o] = Xt[p,:] . Wq[o,:] + b_qkv[o]; one wave computes a 32x32 tile.
__global__ void __launch_bounds__(128)
qkv_gemm_kernel(const u16* __restrict__ Xt, const u16* __restrict__ Wq,
                const float* __restrict__ bqkv,
                u16* __restrict__ Qb, u16* __restrict__ Kb, u16* __restrict__ Vt) {
    int lane = threadIdx.x & 31;
    int wid  = threadIdx.x >> 5;
    int t = blockIdx.x * 4 + wid;                  // 4*128*24 = 12288 waves
    int nt = t % 24;
    int rest = t / 24;
    int mt = rest & 127;
    int b  = rest >> 7;
    int m = lane & 15, kh = lane >> 4;
    int mbase = mt * 32, nbase = nt * 32;

    const u16* Arow = Xt + (size_t)b * 4096 * 256;
    const u16* pa0 = Arow + (size_t)(mbase + m) * 256 + kh * 8;
    const u16* pa1 = pa0 + 16 * 256;
    const u16* pb0 = Wq + (size_t)(nbase + m) * 256 + kh * 8;
    const u16* pb1 = pb0 + 16 * 256;

    v8f a00 = vzero8(), a01 = vzero8(), a10 = vzero8(), a11 = vzero8();
#pragma unroll 2
    for (int k0 = 0; k0 < 256; k0 += 32) {
        v16bf af0 = load_a_ptr(pa0); pa0 += 32;
        v16bf af1 = load_a_ptr(pa1); pa1 += 32;
        v16bf bf0 = load_b_ptr(pb0); pb0 += 32;
        v16bf bf1 = load_b_ptr(pb1); pb1 += 32;
        a00 = wmma_bf16(af0, bf0, a00);
        a01 = wmma_bf16(af0, bf1, a01);
        a10 = wmma_bf16(af1, bf0, a10);
        a11 = wmma_bf16(af1, bf1, a11);
    }
    qkv_store_tile(a00, bqkv, b, mbase,      nbase,      m, kh, Qb, Kb, Vt);
    qkv_store_tile(a01, bqkv, b, mbase,      nbase + 16, m, kh, Qb, Kb, Vt);
    qkv_store_tile(a10, bqkv, b, mbase + 16, nbase,      m, kh, Qb, Kb, Vt);
    qkv_store_tile(a11, bqkv, b, mbase + 16, nbase + 16, m, kh, Qb, Kb, Vt);
}

// ---------------- flash attention (bf16 WMMA, f32 stats) ----------------
__global__ void __launch_bounds__(128)
attn_kernel(const u16* __restrict__ Qb, const u16* __restrict__ Kb,
            const u16* __restrict__ Vt, u16* __restrict__ Ab) {
    __shared__ __align__(16) u16 Plds[4][16][32];  // per-wave P scratch
    int lane = threadIdx.x & 31;
    int wid  = threadIdx.x >> 5;
    int wgid = blockIdx.x * 4 + wid;               // 8192 waves
    int mt = wgid & 255;                           // q-row tile
    int bh = wgid >> 8;                            // 0..31
    int b = bh >> 3, head = bh & 7;
    int m = lane & 15, kh = lane >> 4;

    const u16* Qrow = Qb + (size_t)bh * 4096 * 32;
    const u16* Krow = Kb + (size_t)bh * S_TOT * 32;
    const u16* Vrow = Vt + (size_t)bh * 32 * S_TOT;

    // Q A-fragment: K dim = head_dim = 32 -> single fragment, loaded once.
    v16bf qf = load_a_ptr(Qrow + (size_t)(mt * 16 + m) * 32 + kh * 8);

    // Per-lane streaming pointers (advance by fixed strides each iteration).
    const u16* pk0 = Krow + (size_t)m * 32 + kh * 8;          // K rows s0..s0+15
    const u16* pk1 = pk0 + 16 * 32;                            // K rows s0+16..s0+31
    const u16* pv0 = Vrow + (size_t)m * S_TOT + kh * 8;       // Vt rows d=0..15
    const u16* pv1 = pv0 + (size_t)16 * S_TOT;                 // Vt rows d=16..31
    const u16* pp  = &Plds[wid][0][0] + (size_t)m * 32 + kh * 8; // P A-frag pointer

    v8f acc0 = vzero8();                           // O columns d=0..15
    v8f acc1 = vzero8();                           // O columns d=16..31
    float mi[8], li[8];
#pragma unroll
    for (int r = 0; r < 8; r++) { mi[r] = -1e30f; li[r] = 0.f; }

    for (int t = 0; t < S_TOT / 32; ++t) {
        // sim = q @ k^T : B[k=d][n=s] = Kb[s][d]
        v16bf kf0 = load_b_ptr(pk0); pk0 += 32 * 32;
        v16bf kf1 = load_b_ptr(pk1); pk1 += 32 * 32;
        v8f c0 = wmma_bf16(qf, kf0, vzero8());
        v8f c1 = wmma_bf16(qf, kf1, vzero8());

        float bias0 = 0.f, bias1 = 0.f;
        if (t == 0) {                              // mask pad slots 4..31
            if (m >= 4) bias0 = -1e30f;
            bias1 = -1e30f;
        }

        float p0[8], p1[8];
#pragma unroll
        for (int r = 0; r < 8; r++) {
            float a  = c0[r] + bias0;
            float bb = c1[r] + bias1;
            float tm = fmaxf(a, bb);
            for (int msk = 8; msk; msk >>= 1)      // row-max across the 16-lane group
                tm = fmaxf(tm, __shfl_xor(tm, msk));
            float nm = fmaxf(mi[r], tm);
            float e0 = __expf(a - nm);
            float e1 = __expf(bb - nm);
            float rs = e0 + e1;
            for (int msk = 8; msk; msk >>= 1)      // row-sum across the 16-lane group
                rs += __shfl_xor(rs, msk);
            float alpha = __expf(mi[r] - nm);
            li[r] = li[r] * alpha + rs;
            mi[r] = nm;
            acc0[r] *= alpha;
            acc1[r] *= alpha;
            p0[r] = e0; p1[r] = e1;
        }

        // C-frag (lane=col) -> A-frag (lane=row) transpose via per-wave LDS
#pragma unroll
        for (int r = 0; r < 8; r++) {
            Plds[wid][kh * 8 + r][m]      = f2bf(p0[r]);
            Plds[wid][kh * 8 + r][16 + m] = f2bf(p1[r]);
        }
        v16bf pf  = load_a_ptr(pp);
        // O += P @ V : B[k=s][n=d] = V[s][d] = Vt[d][s]
        v16bf vf0 = load_b_ptr(pv0); pv0 += 32;
        v16bf vf1 = load_b_ptr(pv1); pv1 += 32;
        acc0 = wmma_bf16(pf, vf0, acc0);
        acc1 = wmma_bf16(pf, vf1, acc1);
    }

#pragma unroll
    for (int r = 0; r < 8; r++) {
        int p = mt * 16 + kh * 8 + r;
        float inv = 1.f / li[r];
        size_t base = ((size_t)(b * 4096 + p)) * 256 + head * 32;
        Ab[base + m]      = f2bf(acc0[r] * inv);
        Ab[base + 16 + m] = f2bf(acc1[r] * inv);
    }
}

// ---------------- output projection GEMM (2x2 register-blocked) ----------------

__device__ __forceinline__ void out_store_tile(v8f acc, const float* __restrict__ bout,
                                               int b, int mbase, int obase, int m, int kh,
                                               float* __restrict__ Out) {
    int o = obase + m;
    float bias = bout[o];
#pragma unroll
    for (int r = 0; r < 8; r++) {
        int p = mbase + kh * 8 + r;
        Out[((size_t)(b * 256 + o)) * 4096 + p] = acc[r] + bias;
    }
}

__global__ void __launch_bounds__(128)
out_gemm_kernel(const u16* __restrict__ Ab, const u16* __restrict__ Wo,
                const float* __restrict__ bout, float* __restrict__ Out) {
    int lane = threadIdx.x & 31;
    int wid  = threadIdx.x >> 5;
    int t = blockIdx.x * 4 + wid;                  // 4*128*8 = 4096 waves
    int nt = t & 7;
    int rest = t >> 3;
    int mt = rest & 127;
    int b  = rest >> 7;
    int m = lane & 15, kh = lane >> 4;
    int mbase = mt * 32, nbase = nt * 32;

    const u16* Arow = Ab + (size_t)b * 4096 * 256;
    const u16* pa0 = Arow + (size_t)(mbase + m) * 256 + kh * 8;
    const u16* pa1 = pa0 + 16 * 256;
    const u16* pb0 = Wo + (size_t)(nbase + m) * 256 + kh * 8;
    const u16* pb1 = pb0 + 16 * 256;

    v8f a00 = vzero8(), a01 = vzero8(), a10 = vzero8(), a11 = vzero8();
#pragma unroll 2
    for (int k0 = 0; k0 < 256; k0 += 32) {
        v16bf af0 = load_a_ptr(pa0); pa0 += 32;
        v16bf af1 = load_a_ptr(pa1); pa1 += 32;
        v16bf bf0 = load_b_ptr(pb0); pb0 += 32;
        v16bf bf1 = load_b_ptr(pb1); pb1 += 32;
        a00 = wmma_bf16(af0, bf0, a00);
        a01 = wmma_bf16(af0, bf1, a01);
        a10 = wmma_bf16(af1, bf0, a10);
        a11 = wmma_bf16(af1, bf1, a11);
    }
    out_store_tile(a00, bout, b, mbase,      nbase,      m, kh, Out);
    out_store_tile(a01, bout, b, mbase,      nbase + 16, m, kh, Out);
    out_store_tile(a10, bout, b, mbase + 16, nbase,      m, kh, Out);
    out_store_tile(a11, bout, b, mbase + 16, nbase + 16, m, kh, Out);
}

// ---------------- launch ----------------

extern "C" void kernel_launch(void* const* d_in, const int* in_sizes, int n_in,
                              void* d_out, int out_size, void* d_ws, size_t ws_size,
                              hipStream_t stream) {
    const float* x      = (const float*)d_in[0];   // (4,256,64,64)
    const float* w_qkv  = (const float*)d_in[1];   // (768,256)
    const float* b_qkv  = (const float*)d_in[2];   // (768,)
    const float* mem_kv = (const float*)d_in[3];   // (2,8,4,32)
    const float* w_out  = (const float*)d_in[4];   // (256,256)
    const float* b_out  = (const float*)d_in[5];   // (256,)
    float* out = (float*)d_out;                    // (4,256,64,64)

    char* ws = (char*)d_ws;
    // workspace layout (bytes, all 16B aligned)
    u16* Xt = (u16*)(ws + 0);                       //  4*4096*256*2 = 8,388,608
    u16* Qb = (u16*)(ws + 8388608);                 // 32*4096*32*2  = 8,388,608
    u16* Kb = (u16*)(ws + 16777216);                // 32*4128*32*2  = 8,454,144
    u16* Vt = (u16*)(ws + 25231360);                // 32*32*4128*2  = 8,454,144
    u16* Ab = (u16*)(ws + 33685504);                //  4*4096*256*2 = 8,388,608
    u16* Wq = (u16*)(ws + 42074112);                // 768*256*2     =   393,216
    u16* Wo = (u16*)(ws + 42467328);                // 256*256*2     =   131,072

    prep_x_kernel<<<16384, 256, 0, stream>>>(x, Xt);
    cvt_bf16_kernel<<<768, 256, 0, stream>>>(w_qkv, Wq, 768 * 256);
    cvt_bf16_kernel<<<256, 256, 0, stream>>>(w_out, Wo, 256 * 256);
    prep_mem_kernel<<<128, 256, 0, stream>>>(mem_kv, Kb, Vt);

    qkv_gemm_kernel<<<3072, 128, 0, stream>>>(Xt, Wq, b_qkv, Qb, Kb, Vt);
    attn_kernel<<<2048, 128, 0, stream>>>(Qb, Kb, Vt, Ab);
    out_gemm_kernel<<<1024, 128, 0, stream>>>(Ab, Wo, b_out, out);
}
